// RoPEAttention_62191126446094
// MI455X (gfx1250) — compile-verified, matrix operand order checked by reading, not run on hardware
//
#include <hip/hip_runtime.h>
#include <hip/hip_bf16.h>
#include <stdint.h>

typedef __attribute__((ext_vector_type(16))) __bf16   v16bf;
typedef __attribute__((ext_vector_type(4)))  __bf16   v4bf;
typedef __attribute__((ext_vector_type(8)))  float    v8f;
typedef __attribute__((ext_vector_type(8)))  uint32_t v8u;

constexpr int Bk = 2, Lk = 2048, Dk = 1024, Hk = 16, HDk = 64;
constexpr int Mtot = Bk * Lk;           // 4096

// ---------------- fp32 -> bf16 convert (x4 vectorized) ----------------
__global__ void cvt_bf16_kernel(const float* __restrict__ in,
                                __bf16* __restrict__ out, int n4) {
  int i = blockIdx.x * blockDim.x + threadIdx.x;
  if (i >= n4) return;
  float4 f = ((const float4*)in)[i];
  v4bf h;
  h[0] = (__bf16)f.x; h[1] = (__bf16)f.y; h[2] = (__bf16)f.z; h[3] = (__bf16)f.w;
  ((v4bf*)out)[i] = h;
}

// Fragment K-offset pattern for 16-bit A/B matrices (ISA 7.12.2):
// vgpr j holds K = 2*(j&3) + 16*(j>>2) + 8*(lane>>4), K+1.
__device__ __forceinline__ int frag_koff(int j, int kh) {
  return 2 * (j & 3) + 16 * (j >> 2) + 8 * kh;
}

// Async global->LDS copy of 16 bytes per lane (CDNA5 ASYNCcnt path, GV mode).
__device__ __forceinline__ void async_b128(uint32_t lds_off, const void* gaddr) {
  asm volatile("global_load_async_to_lds_b128 %0, %1, off"
               :: "v"(lds_off), "v"((uint64_t)(uintptr_t)gaddr) : "memory");
}
__device__ __forceinline__ void wait_async0() {
  asm volatile("s_wait_asynccnt 0x0" ::: "memory");
}
// Low 32 bits of a generic LDS pointer == LDS byte offset (flat aperture rule).
__device__ __forceinline__ uint32_t lds_off32(const void* p) {
  return (uint32_t)(uintptr_t)p;
}

// ---------------- tiled bf16 WMMA GEMM: out = A[M,K] * B[N,K]^T ----------------
// mode 0: bf16 out to [B,H,L,HD] (V path)
// mode 1: bf16 out to [B,H,L,HD] with RoPE (Q/K path), scale folded in
// mode 2: fp32 out to [M,N] (final projection)
constexpr int BM = 128, BN = 128, BKg = 64;
constexpr int LDA = BKg + 8;            // 72 shorts: 144B row stride (16B aligned, conflict-free)
constexpr int TILE_SH = BM * LDA;       // elements per buffer

__global__ __launch_bounds__(256) void gemm_wmma_kernel(
    const __bf16* __restrict__ A, const __bf16* __restrict__ Bm,
    void* __restrict__ outp, int M, int N, int K, int mode, float scale) {
  __shared__ __bf16 shA[2][TILE_SH];
  __shared__ __bf16 shB[2][TILE_SH];

  const int tid  = threadIdx.x;
  const int lane = tid & 31;
  const int wave = tid >> 5;
  const int mTile = blockIdx.y * BM;
  const int nTile = blockIdx.x * BN;
  const int wm = (wave >> 1) * 32;      // wave row offset: 0/32/64/96
  const int wn = (wave & 1) * 64;       // wave col offset: 0/64
  const int lm = lane & 15;
  const int kh = lane >> 4;

  const uint32_t aBase = lds_off32(&shA[0][0]);
  const uint32_t bBase = lds_off32(&shB[0][0]);

  v8f acc[2][4] = {};

  // async-issue one 128x64 bf16 tile pair into buffer `buf` for k-slice k0
  auto issue = [&](int buf, int k0) {
    const uint32_t bufOff = (uint32_t)buf * (TILE_SH * 2);
#pragma unroll
    for (int i = 0; i < 4; ++i) {
      int idx = tid + i * 256;
      int r = idx >> 3, c = (idx & 7) * 8;
      uint32_t loff = (uint32_t)((r * LDA + c) * 2) + bufOff;
      async_b128(aBase + loff, A + (size_t)(mTile + r) * K + k0 + c);
      async_b128(bBase + loff, Bm + (size_t)(nTile + r) * K + k0 + c);
    }
  };

  issue(0, 0);
  int buf = 0;
  for (int k0 = 0; k0 < K; k0 += BKg) {
    wait_async0();                      // my buffer-`buf` copies done
    __syncthreads();                    // everyone's copies done / prev reads done
    if (k0 + BKg < K) issue(buf ^ 1, k0 + BKg);

    const __bf16* sA = &shA[buf][0];
    const __bf16* sB = &shB[buf][0];
#pragma unroll
    for (int s = 0; s < 2; ++s) {       // two K=32 sub-steps
      const int ks = s * 32;
      v16bf af[2];
#pragma unroll
      for (int mf = 0; mf < 2; ++mf) {
        v8u t;
        const int row = wm + mf * 16 + lm;
#pragma unroll
        for (int j = 0; j < 8; ++j)
          t[j] = *(const uint32_t*)(&sA[row * LDA + ks + frag_koff(j, kh)]);
        af[mf] = __builtin_bit_cast(v16bf, t);
      }
#pragma unroll
      for (int nf = 0; nf < 4; ++nf) {
        v8u t;
        const int row = wn + nf * 16 + lm;
#pragma unroll
        for (int j = 0; j < 8; ++j)
          t[j] = *(const uint32_t*)(&sB[row * LDA + ks + frag_koff(j, kh)]);
        v16bf bf = __builtin_bit_cast(v16bf, t);
#pragma unroll
        for (int mf = 0; mf < 2; ++mf)
          acc[mf][nf] = __builtin_amdgcn_wmma_f32_16x16x32_bf16(
              false, af[mf], false, bf, (short)0, acc[mf][nf], false, false);
      }
    }
    buf ^= 1;
  }

  // ---------------- epilogue ----------------
  if (mode == 2) {                      // fp32 [M,N]
    float* o = (float*)outp;
#pragma unroll
    for (int mf = 0; mf < 2; ++mf)
#pragma unroll
      for (int nf = 0; nf < 4; ++nf)
#pragma unroll
        for (int r = 0; r < 8; ++r) {
          int m = mTile + wm + mf * 16 + kh * 8 + r;
          int n = nTile + wn + nf * 16 + lm;
          o[(size_t)m * N + n] = acc[mf][nf][r];
        }
    return;
  }

  // bf16 [B,H,L,HD]; wave's 64-col span == exactly one head (64-aligned)
  __bf16* o = (__bf16*)outp;
#pragma unroll
  for (int mf = 0; mf < 2; ++mf)
#pragma unroll
    for (int r = 0; r < 8; ++r) {
      const int m = mTile + wm + mf * 16 + kh * 8 + r;
      const int b = m >> 11;            // /L
      const int l = m & (Lk - 1);
#pragma unroll
      for (int nf = 0; nf < 4; ++nf) {
        const int n = nTile + wn + nf * 16 + lm;
        const int h = n >> 6;
        const int d = n & 63;
        float v = acc[mf][nf][r];
        if (mode == 1) {                // RoPE: pair lives in frag nf^2, same lane
          const float p = acc[mf][nf ^ 2][r];
          const int dm = d & 31;
          // inv_freq = 10000^(-dm/32) = exp(-dm * ln(1e4)/32)
          const float ang = (float)l * __expf(-(float)dm * 0.28782313662f);
          const float c = __cosf(ang), s2 = __sinf(ang);
          v = (nf < 2) ? (v * c - p * s2) : (v * c + p * s2);
        }
        v *= scale;
        o[(((size_t)b * Hk + h) * Lk + l) * HDk + d] = (__bf16)v;
      }
    }
}

// ---------------- flash attention: O = softmax(Q K^T) V per (b,h) ----------------
constexpr int QB = 64, KBl = 64, LDK = 72;

__global__ __launch_bounds__(128) void flash_wmma_kernel(
    const __bf16* __restrict__ Q, const __bf16* __restrict__ Km,
    const __bf16* __restrict__ V, __bf16* __restrict__ O) {
  __shared__ __bf16 shK[KBl * LDK];          // [key][d]
  __shared__ __bf16 shV[HDk * LDK];          // transposed: [d][key]
  __shared__ __bf16 shP[4][16 * LDK];        // per-wave P tile

  const int tid  = threadIdx.x;
  const int lane = tid & 31;
  const int wave = tid >> 5;
  const int bh = blockIdx.y;
  const int b = bh / Hk, h = bh % Hk;
  const int q0 = blockIdx.x * QB + wave * 16;
  const int lm = lane & 15, kh = lane >> 4;

  const __bf16* qb = Q + (size_t)bh * Lk * HDk;
  const __bf16* kb = Km + (size_t)bh * Lk * HDk;
  const __bf16* vb = V + (size_t)bh * Lk * HDk;
  const uint32_t kBase = lds_off32(&shK[0]);

  // register-resident Q fragments (16 rows x 64 d -> 2 A-frags)
  v16bf qa[2];
#pragma unroll
  for (int s = 0; s < 2; ++s) {
    v8u t;
    const int row = q0 + lm;
#pragma unroll
    for (int j = 0; j < 8; ++j)
      t[j] = *(const uint32_t*)(qb + (size_t)row * HDk + s * 32 + frag_koff(j, kh));
    qa[s] = __builtin_bit_cast(v16bf, t);
  }

  v8f oacc[4] = {};
  float mrow[8], lrow[8];
#pragma unroll
  for (int r = 0; r < 8; ++r) { mrow[r] = -3.0e38f; lrow[r] = 0.f; }

  for (int kv0 = 0; kv0 < Lk; kv0 += KBl) {
    __syncthreads();                         // prev reads of shK/shV done
#pragma unroll
    for (int i = 0; i < 4; ++i) {            // K async; V through VGPRs (transpose)
      int idx = tid + i * 128;
      int r = idx >> 3, c = (idx & 7) * 8;
      async_b128(kBase + (uint32_t)((r * LDK + c) * 2),
                 kb + (size_t)(kv0 + r) * HDk + c);
      uint4 vv = *(const uint4*)(vb + (size_t)(kv0 + r) * HDk + c);
      __bf16 tmp[8];
      *(uint4*)tmp = vv;
#pragma unroll
      for (int j = 0; j < 8; ++j) shV[(c + j) * LDK + r] = tmp[j];
    }
    wait_async0();
    __syncthreads();

    // S = Q K^T  (4 n-frags over 64 keys, 2 K=32 sub-steps)
    v8f sacc[4] = {};
#pragma unroll
    for (int s = 0; s < 2; ++s)
#pragma unroll
      for (int nf = 0; nf < 4; ++nf) {
        v8u t;
        const int row = nf * 16 + lm;
#pragma unroll
        for (int j = 0; j < 8; ++j)
          t[j] = *(const uint32_t*)(&shK[row * LDK + s * 32 + frag_koff(j, kh)]);
        v16bf bfrag = __builtin_bit_cast(v16bf, t);
        sacc[nf] = __builtin_amdgcn_wmma_f32_16x16x32_bf16(
            false, qa[s], false, bfrag, (short)0, sacc[nf], false, false);
      }

    // online softmax: row m = kh*8 + r lives in one 16-lane group
    float alpha[8];
#pragma unroll
    for (int r = 0; r < 8; ++r) {
      float mx = fmaxf(fmaxf(sacc[0][r], sacc[1][r]), fmaxf(sacc[2][r], sacc[3][r]));
#pragma unroll
      for (int off = 8; off >= 1; off >>= 1) mx = fmaxf(mx, __shfl_xor(mx, off, 32));
      const float mnew = fmaxf(mrow[r], mx);
      alpha[r] = __expf(mrow[r] - mnew);
      float sum = 0.f;
#pragma unroll
      for (int nf = 0; nf < 4; ++nf) {
        const float e = __expf(sacc[nf][r] - mnew);
        sacc[nf][r] = e;
        sum += e;
      }
#pragma unroll
      for (int off = 8; off >= 1; off >>= 1) sum += __shfl_xor(sum, off, 32);
      lrow[r] = lrow[r] * alpha[r] + sum;
      mrow[r] = mnew;
    }

    // transpose P through per-wave LDS into A-fragment layout
    __bf16* pw = shP[wave];
#pragma unroll
    for (int nf = 0; nf < 4; ++nf)
#pragma unroll
      for (int r = 0; r < 8; ++r)
        pw[(kh * 8 + r) * LDK + nf * 16 + lm] = (__bf16)sacc[nf][r];
    __syncthreads();

    v16bf pa[2];
#pragma unroll
    for (int s = 0; s < 2; ++s) {
      v8u t;
#pragma unroll
      for (int j = 0; j < 8; ++j)
        t[j] = *(const uint32_t*)(&pw[lm * LDK + s * 32 + frag_koff(j, kh)]);
      pa[s] = __builtin_bit_cast(v16bf, t);
    }

    // O = O*alpha + P V
#pragma unroll
    for (int f = 0; f < 4; ++f) {
#pragma unroll
      for (int r = 0; r < 8; ++r) oacc[f][r] *= alpha[r];
#pragma unroll
      for (int s = 0; s < 2; ++s) {
        v8u t;
        const int dcol = f * 16 + lm;
#pragma unroll
        for (int j = 0; j < 8; ++j)
          t[j] = *(const uint32_t*)(&shV[dcol * LDK + s * 32 + frag_koff(j, kh)]);
        v16bf vfrag = __builtin_bit_cast(v16bf, t);
        oacc[f] = __builtin_amdgcn_wmma_f32_16x16x32_bf16(
            false, pa[s], false, vfrag, (short)0, oacc[f], false, false);
      }
    }
  }

  // normalize + store to [B, L, D] bf16 for the final projection
#pragma unroll
  for (int f = 0; f < 4; ++f)
#pragma unroll
    for (int r = 0; r < 8; ++r) {
      const int qrow = q0 + kh * 8 + r;
      const int d = f * 16 + lm;
      const float v = oacc[f][r] / lrow[r];
      O[((size_t)b * Lk + qrow) * Dk + h * HDk + d] = (__bf16)v;
    }
}

// ---------------- host launch ----------------
extern "C" void kernel_launch(void* const* d_in, const int* in_sizes, int n_in,
                              void* d_out, int out_size, void* d_ws, size_t ws_size,
                              hipStream_t stream) {
  const float* x  = (const float*)d_in[0];
  const float* wq = (const float*)d_in[1];
  const float* wk = (const float*)d_in[2];
  const float* wv = (const float*)d_in[3];
  const float* wo = (const float*)d_in[4];
  float* out = (float*)d_out;

  char* ws = (char*)d_ws;
  const size_t MiB = 1024 * 1024;
  __bf16* xb   = (__bf16*)(ws + 0 * MiB);   // 8 MiB  [4096,1024]
  __bf16* wqb  = (__bf16*)(ws + 8 * MiB);   // 2 MiB
  __bf16* wkb  = (__bf16*)(ws + 10 * MiB);
  __bf16* wvb  = (__bf16*)(ws + 12 * MiB);
  __bf16* wob  = (__bf16*)(ws + 14 * MiB);
  __bf16* qbuf = (__bf16*)(ws + 16 * MiB);  // 8 MiB [B,H,L,HD]
  __bf16* kbuf = (__bf16*)(ws + 24 * MiB);
  __bf16* vbuf = (__bf16*)(ws + 32 * MiB);
  __bf16* obuf = (__bf16*)(ws + 40 * MiB);  // 8 MiB [B,L,D]

  const int nx4 = Mtot * Dk / 4;            // 1048576
  const int nw4 = Dk * Dk / 4;              // 262144
  cvt_bf16_kernel<<<nx4 / 256, 256, 0, stream>>>(x, xb, nx4);
  cvt_bf16_kernel<<<nw4 / 256, 256, 0, stream>>>(wq, wqb, nw4);
  cvt_bf16_kernel<<<nw4 / 256, 256, 0, stream>>>(wk, wkb, nw4);
  cvt_bf16_kernel<<<nw4 / 256, 256, 0, stream>>>(wv, wvb, nw4);
  cvt_bf16_kernel<<<nw4 / 256, 256, 0, stream>>>(wo, wob, nw4);

  dim3 gg(Dk / BN, Mtot / BM);              // (8, 32)
  // Q: RoPE + softmax scale folded in; K: RoPE; V: plain
  gemm_wmma_kernel<<<gg, 256, 0, stream>>>(xb, wqb, qbuf, Mtot, Dk, Dk, 1, 0.125f);
  gemm_wmma_kernel<<<gg, 256, 0, stream>>>(xb, wkb, kbuf, Mtot, Dk, Dk, 1, 1.0f);
  gemm_wmma_kernel<<<gg, 256, 0, stream>>>(xb, wvb, vbuf, Mtot, Dk, Dk, 0, 1.0f);

  dim3 fg(Lk / QB, Bk * Hk);                // (32, 32)
  flash_wmma_kernel<<<fg, 128, 0, stream>>>(qbuf, kbuf, vbuf, obuf);

  gemm_wmma_kernel<<<gg, 256, 0, stream>>>(obuf, wob, out, Mtot, Dk, Dk, 2, 1.0f);
}
